// Attention_76845554860188
// MI455X (gfx1250) — matrix operand order repaired and checked by reading, not compiled, not run on hardware
//
#include <hip/hip_runtime.h>

// ---------------------------------------------------------------------------
// Problem constants (match reference)
// ---------------------------------------------------------------------------
constexpr int SLEN   = 2048;
constexpr int DMODEL = 1024;
constexpr int NHEAD  = 16;
constexpr int NKV    = 4;
constexpr int HDIM   = 64;

// ---------------------------------------------------------------------------
// Types for CDNA5 WMMA (wave32): bf16 16x16x32 -> f32 16x16
// ---------------------------------------------------------------------------
typedef __attribute__((ext_vector_type(16))) __bf16 bf16x16;
typedef __attribute__((ext_vector_type(8)))  float  f32x8;
typedef __attribute__((ext_vector_type(4)))  unsigned int u32x4;

union Frag16 {
    bf16x16        v;
    u32x4          q[2];
    unsigned short s[16];
};

__device__ __forceinline__ unsigned short f2bf(float f) {
    union { float f; unsigned u; } cv; cv.f = f;
    unsigned r = cv.u + 0x7FFFu + ((cv.u >> 16) & 1u);   // round-to-nearest-even
    return (unsigned short)(r >> 16);
}

// A fragment (16x32, 16-bit): lane ln = row; lanes 0-15 hold K {0..7,16..23},
// lanes 16-31 hold K {8..15,24..31}.  Two global b128 loads.
__device__ __forceinline__ Frag16 load_a_frag(const unsigned short* p, int half) {
    Frag16 a;
    a.q[0] = *(const u32x4*)(p + half * 8);
    a.q[1] = *(const u32x4*)(p + half * 8 + 16);
    return a;
}
// B fragment (32x16, 16-bit) from a K-contiguous (transposed) source row:
// lane = column, lanes 0-15 hold K {0..7,16..23}, lanes 16-31 hold K {8..15,24..31}.
__device__ __forceinline__ Frag16 load_b_frag(const unsigned short* p, int half) {
    Frag16 b;
    b.q[0] = *(const u32x4*)(p + half * 8);
    b.q[1] = *(const u32x4*)(p + half * 8 + 16);
    return b;
}

// ---------------------------------------------------------------------------
// fp32 -> bf16 conversion (plain)
// ---------------------------------------------------------------------------
__global__ void cvt_bf16_kernel(const float* __restrict__ src,
                                unsigned short* __restrict__ dst, int n) {
    int i = blockIdx.x * blockDim.x + threadIdx.x;
    if (i < n) dst[i] = f2bf(src[i]);
}

// fp32 [Kd x N] row-major -> bf16 [N x Kd] (transposed), so WMMA B fragments
// become K-contiguous per-lane b128 loads.
__global__ void cvt_transpose_kernel(const float* __restrict__ src,
                                     unsigned short* __restrict__ dst,
                                     int Kd, int N) {
    int i = blockIdx.x * blockDim.x + threadIdx.x;
    if (i < Kd * N) {
        int k = i / N, n = i % N;
        dst[(size_t)n * Kd + k] = f2bf(src[i]);
    }
}

// ---------------------------------------------------------------------------
// QKV projection + RoPE.  One wave per (matrix, head, 16-row block):
// items: 2048 Q + 512 K + 512 V = 3072 -> 768 blocks x 128 threads (4 waves).
// Each wave computes a 16x64 tile (one head) as 4 C fragments, reusing one A
// fragment across 4 WMMAs per 32-wide k-step.  No LDS, no barriers.
// RoPE pairs (d, d+32) are C tiles t and t+2 at the same lane/row -> applied
// entirely in registers.
// Q -> Qbuf[h][s][64], K -> Kbuf[kv][s][64], V -> Vtbuf[kv][64][s] (transposed)
// ---------------------------------------------------------------------------
__global__ __launch_bounds__(128) void qkv_rope_kernel(
    const unsigned short* __restrict__ xb,
    const unsigned short* __restrict__ Wqt,   // [1024][1024] (N x K)
    const unsigned short* __restrict__ Wkt,   // [256][1024]
    const unsigned short* __restrict__ Wvt,   // [256][1024]
    const float* __restrict__ cosT, const float* __restrict__ sinT,
    unsigned short* __restrict__ Qbuf,
    unsigned short* __restrict__ Kbuf,
    unsigned short* __restrict__ Vtbuf)
{
    const int tid  = threadIdx.x;
    const int wv   = tid >> 5;
    const int lane = tid & 31;
    const int half = lane >> 4;
    const int ln   = lane & 15;

    const int item = blockIdx.x * 4 + wv;     // 0..3071
    int mode, head, rowblk;
    if (item < 2048)      { mode = 0; head = item >> 7;          rowblk = item & 127; }
    else if (item < 2560) { mode = 1; head = (item - 2048) >> 7; rowblk = (item - 2048) & 127; }
    else                  { mode = 2; head = (item - 2560) >> 7; rowblk = (item - 2560) & 127; }

    const unsigned short* Bt = (mode == 0) ? Wqt : ((mode == 1) ? Wkt : Wvt);
    const int col0 = head * HDIM;
    const int row0 = rowblk * 16;

    f32x8 acc[4] = {};
    const unsigned short* arow = xb + (size_t)(row0 + ln) * DMODEL;

    for (int k0 = 0; k0 < DMODEL; k0 += 32) {
        Frag16 a = load_a_frag(arow + k0, half);
        if (k0 + 32 < DMODEL)
            __builtin_prefetch(arow + k0 + 32, 0, 1);
        #pragma unroll
        for (int t = 0; t < 4; ++t) {
            const unsigned short* bp =
                Bt + (size_t)(col0 + t * 16 + ln) * DMODEL + k0;
            Frag16 b = load_b_frag(bp, half);
            acc[t] = __builtin_amdgcn_wmma_f32_16x16x32_bf16(
                         false, a.v, false, b.v, (short)0, acc[t], false, false);
        }
    }

    if (mode <= 1) {
        unsigned short* Out = (mode == 0)
            ? (Qbuf + (size_t)head * SLEN * HDIM)
            : (Kbuf + (size_t)head * SLEN * HDIM);
        #pragma unroll
        for (int t = 0; t < 2; ++t) {
            const int d = t * 16 + ln;                // 0..31
            #pragma unroll
            for (int r = 0; r < 8; ++r) {
                const int srow = row0 + r + half * 8;
                const float c  = cosT[srow * 32 + d];
                const float sn = sinT[srow * 32 + d];
                const float t1 = acc[t][r];
                const float t2 = acc[t + 2][r];
                Out[(size_t)srow * HDIM + d]      = f2bf(t1 * c - t2 * sn);
                Out[(size_t)srow * HDIM + d + 32] = f2bf(t2 * c + t1 * sn);
            }
        }
    } else {
        unsigned short* Out = Vtbuf + (size_t)head * HDIM * SLEN;  // [d][s]
        #pragma unroll
        for (int t = 0; t < 4; ++t) {
            const int d = t * 16 + ln;
            #pragma unroll
            for (int r = 0; r < 8; ++r)
                Out[(size_t)d * SLEN + row0 + r + half * 8] = f2bf(acc[t][r]);
        }
    }
}

// ---------------------------------------------------------------------------
// Flash attention: one wave (32 threads) per (head, 16-query-row block).
// grid.x = 16 * 128 = 2048.  Key blocks of 32; online softmax in f32;
// P relayouted C-format -> A-format through LDS.
// ---------------------------------------------------------------------------
__global__ __launch_bounds__(32) void attn_kernel(
    const unsigned short* __restrict__ Qbuf,
    const unsigned short* __restrict__ Kbuf,
    const unsigned short* __restrict__ Vtbuf,
    unsigned short* __restrict__ AO)
{
    __shared__ unsigned short Plds[16][32];

    const int h    = blockIdx.x >> 7;
    const int qblk = blockIdx.x & 127;
    const int kv   = h >> 2;              // N_REP = 4
    const int q0   = qblk * 16;
    const int lane = threadIdx.x & 31;
    const int half = lane >> 4;
    const int ln   = lane & 15;

    // Q fragments (A-matrix), loop-invariant: d-steps 0..31 and 32..63
    Frag16 aq[2];
    const unsigned short* qp = Qbuf + ((size_t)h * SLEN + q0 + ln) * HDIM;
    #pragma unroll
    for (int d2 = 0; d2 < 2; ++d2)
        aq[d2] = load_a_frag(qp + d2 * 32, half);

    f32x8 O[4] = {};
    float m[8], l[8];
    #pragma unroll
    for (int r = 0; r < 8; ++r) { m[r] = -1e30f; l[r] = 0.0f; }

    const float scale = 0.125f;                 // HD^-0.5
    const float L2E   = 1.4426950408889634f;

    for (int key0 = 0; key0 <= q0 + 15; key0 += 32) {
        // ---- scores: S[16q x 32k] = Q Kt, 4 WMMAs ----
        f32x8 sc[2] = {{}, {}};
        #pragma unroll
        for (int kt = 0; kt < 2; ++kt) {
            const unsigned short* kp =
                Kbuf + ((size_t)kv * SLEN + key0 + kt * 16 + ln) * HDIM;
            #pragma unroll
            for (int d2 = 0; d2 < 2; ++d2) {
                Frag16 bk = load_b_frag(kp + d2 * 32, half);
                sc[kt] = __builtin_amdgcn_wmma_f32_16x16x32_bf16(
                             false, aq[d2].v, false, bk.v, (short)0, sc[kt],
                             false, false);
            }
        }

        // ---- causal mask + online softmax (rows split by lane half) ----
        float pr0[8], pr1[8];
        #pragma unroll
        for (int r = 0; r < 8; ++r) {
            const int qrow = q0 + r + half * 8;
            float s0 = sc[0][r] * scale; if (key0 + ln      > qrow) s0 = -1e30f;
            float s1 = sc[1][r] * scale; if (key0 + 16 + ln > qrow) s1 = -1e30f;
            float mx = fmaxf(s0, s1);
            mx = fmaxf(mx, __shfl_xor(mx, 1, 32));
            mx = fmaxf(mx, __shfl_xor(mx, 2, 32));
            mx = fmaxf(mx, __shfl_xor(mx, 4, 32));
            mx = fmaxf(mx, __shfl_xor(mx, 8, 32));
            const float mn   = fmaxf(m[r], mx);
            const float corr = exp2f((m[r] - mn) * L2E);
            const float p0   = exp2f((s0 - mn) * L2E);
            const float p1   = exp2f((s1 - mn) * L2E);
            float rs = p0 + p1;
            rs += __shfl_xor(rs, 1, 32);
            rs += __shfl_xor(rs, 2, 32);
            rs += __shfl_xor(rs, 4, 32);
            rs += __shfl_xor(rs, 8, 32);
            l[r] = l[r] * corr + rs;
            m[r] = mn;
            #pragma unroll
            for (int t = 0; t < 4; ++t) O[t][r] *= corr;
            pr0[r] = p0; pr1[r] = p1;
        }

        // ---- P: C-layout -> A-layout via LDS ----
        __syncthreads();
        #pragma unroll
        for (int r = 0; r < 8; ++r) {
            Plds[r + half * 8][ln]      = f2bf(pr0[r]);
            Plds[r + half * 8][16 + ln] = f2bf(pr1[r]);
        }
        __syncthreads();
        Frag16 ap;
        ap.q[0] = *(const u32x4*)(&Plds[ln][half * 8]);
        ap.q[1] = *(const u32x4*)(&Plds[ln][half * 8 + 16]);

        // ---- O += P V, 4 WMMAs (d-tiles of 16) ----
        #pragma unroll
        for (int t = 0; t < 4; ++t) {
            const unsigned short* vp =
                Vtbuf + ((size_t)kv * HDIM + t * 16 + ln) * SLEN + key0;
            Frag16 bv = load_b_frag(vp, half);
            O[t] = __builtin_amdgcn_wmma_f32_16x16x32_bf16(
                       false, ap.v, false, bv.v, (short)0, O[t], false, false);
        }
    }

    // ---- normalize and store to AO[s][h*64 + d] (bf16) ----
    #pragma unroll
    for (int r = 0; r < 8; ++r) {
        const float inv  = 1.0f / l[r];
        const int   srow = q0 + r + half * 8;
        #pragma unroll
        for (int t = 0; t < 4; ++t)
            AO[(size_t)srow * (NHEAD * HDIM) + h * HDIM + t * 16 + ln] =
                f2bf(O[t][r] * inv);
    }
}

// ---------------------------------------------------------------------------
// Output projection: out[2048x1024] f32 = AO[2048x1024]bf16 @ Wo (Wot is
// pre-transposed [N][K]).  One wave per 16x64 tile: items = 128 rowblks *
// 16 colgroups = 2048 -> 512 blocks x 128 threads.  No LDS, no barriers.
// ---------------------------------------------------------------------------
__global__ __launch_bounds__(128) void out_proj_kernel(
    const unsigned short* __restrict__ AO,
    const unsigned short* __restrict__ Wot,   // [1024][1024] (N x K)
    float* __restrict__ out)
{
    const int tid  = threadIdx.x;
    const int wv   = tid >> 5;
    const int lane = tid & 31;
    const int half = lane >> 4;
    const int ln   = lane & 15;

    const int item   = blockIdx.x * 4 + wv;   // 0..2047
    const int rowblk = item >> 4;
    const int col0   = (item & 15) * 64;
    const int row0   = rowblk * 16;

    f32x8 acc[4] = {};
    const unsigned short* arow = AO + (size_t)(row0 + ln) * DMODEL;

    for (int k0 = 0; k0 < DMODEL; k0 += 32) {
        Frag16 a = load_a_frag(arow + k0, half);
        if (k0 + 32 < DMODEL)
            __builtin_prefetch(arow + k0 + 32, 0, 1);
        #pragma unroll
        for (int t = 0; t < 4; ++t) {
            const unsigned short* bp =
                Wot + (size_t)(col0 + t * 16 + ln) * DMODEL + k0;
            Frag16 b = load_b_frag(bp, half);
            acc[t] = __builtin_amdgcn_wmma_f32_16x16x32_bf16(
                         false, a.v, false, b.v, (short)0, acc[t], false, false);
        }
    }

    #pragma unroll
    for (int r = 0; r < 8; ++r)
        #pragma unroll
        for (int t = 0; t < 4; ++t)
            out[(size_t)(row0 + r + half * 8) * DMODEL + col0 + t * 16 + ln] =
                acc[t][r];
}

// ---------------------------------------------------------------------------
// Host-side launcher
// ---------------------------------------------------------------------------
extern "C" void kernel_launch(void* const* d_in, const int* in_sizes, int n_in,
                              void* d_out, int out_size, void* d_ws, size_t ws_size,
                              hipStream_t stream)
{
    const float* x    = (const float*)d_in[0];
    const float* cosT = (const float*)d_in[1];
    const float* sinT = (const float*)d_in[2];
    const float* Wq   = (const float*)d_in[3];
    const float* Wk   = (const float*)d_in[4];
    const float* Wv   = (const float*)d_in[5];
    const float* Wo   = (const float*)d_in[6];
    float* out = (float*)d_out;

    unsigned short* ws = (unsigned short*)d_ws;
    size_t off = 0;
    unsigned short* xb   = ws + off; off += (size_t)SLEN * DMODEL;
    unsigned short* Wqt  = ws + off; off += (size_t)DMODEL * NHEAD * HDIM;
    unsigned short* Wkt  = ws + off; off += (size_t)DMODEL * NKV * HDIM;
    unsigned short* Wvt  = ws + off; off += (size_t)DMODEL * NKV * HDIM;
    unsigned short* Wot  = ws + off; off += (size_t)(NHEAD * HDIM) * DMODEL;
    unsigned short* Qbuf = ws + off; off += (size_t)NHEAD * SLEN * HDIM;
    unsigned short* Kbuf = ws + off; off += (size_t)NKV * SLEN * HDIM;
    unsigned short* Vtb  = ws + off; off += (size_t)NKV * HDIM * SLEN;
    unsigned short* AO   = ws + off; off += (size_t)SLEN * NHEAD * HDIM;

    // x: plain convert
    {
        int n = SLEN * DMODEL;
        cvt_bf16_kernel<<<(n + 255) / 256, 256, 0, stream>>>(x, xb, n);
    }
    // weights: convert + transpose to [N][K]
    auto cvtT = [&](const float* s, unsigned short* d, int Kd, int N) {
        int n = Kd * N;
        cvt_transpose_kernel<<<(n + 255) / 256, 256, 0, stream>>>(s, d, Kd, N);
    };
    cvtT(Wq, Wqt, DMODEL, NHEAD * HDIM);
    cvtT(Wk, Wkt, DMODEL, NKV * HDIM);
    cvtT(Wv, Wvt, DMODEL, NKV * HDIM);
    cvtT(Wo, Wot, NHEAD * HDIM, DMODEL);

    // QKV + RoPE: 3072 wave-items / 4 waves per block
    qkv_rope_kernel<<<(2048 + 512 + 512) / 4, 128, 0, stream>>>(
        xb, Wqt, Wkt, Wvt, cosT, sinT, Qbuf, Kbuf, Vtb);

    // Attention: 16 heads * 128 query blocks, one wave each
    attn_kernel<<<NHEAD * (SLEN / 16), 32, 0, stream>>>(Qbuf, Kbuf, Vtb, AO);

    // Output projection: 2048 wave-items / 4 waves per block
    out_proj_kernel<<<2048 / 4, 128, 0, stream>>>(AO, Wot, out);
}
